// VectorAttention_45337674776588
// MI455X (gfx1250) — matrix operand order repaired and checked by reading, not compiled, not run on hardware
//
#include <hip/hip_runtime.h>
#include <hip/hip_bf16.h>

typedef __attribute__((ext_vector_type(16))) __bf16 v16bf;
typedef __attribute__((ext_vector_type(8)))  float  v8f;
typedef __attribute__((ext_vector_type(4)))  int    v4i;

#define BB   2
#define NN   1024
#define CH   64
#define TJ   16
#define WAVES 8

#define AS1 __attribute__((address_space(1)))
#define AS3 __attribute__((address_space(3)))

#if defined(__gfx1250__) && __has_builtin(__builtin_amdgcn_global_load_async_to_lds_b128)
#define HAVE_ASYNC_LDS 1
#else
#define HAVE_ASYNC_LDS 0
#endif

#if HAVE_ASYNC_LDS
__device__ __forceinline__ void async_copy16(const float* gsrc, float* ldst) {
    __builtin_amdgcn_global_load_async_to_lds_b128(
        (AS1 v4i*)gsrc, (AS3 v4i*)ldst, 0, 0);
}
template<int N>
__device__ __forceinline__ void wait_asynccnt() {
#if __has_builtin(__builtin_amdgcn_s_wait_asynccnt)
    __builtin_amdgcn_s_wait_asynccnt(N);
#else
    asm volatile("s_wait_asynccnt %0" :: "i"(N));
#endif
}
#endif

__device__ __forceinline__ unsigned short f2bf(float f) {
    union { float f; unsigned u; } x; x.f = f;
    unsigned u = x.u;
    unsigned r = u + 0x7FFFu + ((u >> 16) & 1u);   // round-to-nearest-even
    return (unsigned short)(r >> 16);
}

// ---------------------------------------------------------------------------
// Prep: per row n, q=relu(LN(feat@Wq^T+bq)), k likewise, v=feat@Wv^T+bv,
// then qW1 = q@W1^T and kW1 = k@W1^T + b1 (distributed first linear).
// ---------------------------------------------------------------------------
__global__ void qkv_prep_kernel(const float* __restrict__ feat,
                                const float* __restrict__ Wq, const float* __restrict__ bq,
                                const float* __restrict__ gq, const float* __restrict__ betaq,
                                const float* __restrict__ Wk, const float* __restrict__ bk,
                                const float* __restrict__ gk, const float* __restrict__ betak,
                                const float* __restrict__ Wv, const float* __restrict__ bv,
                                const float* __restrict__ W1, const float* __restrict__ b1,
                                float* __restrict__ qW1, float* __restrict__ kW1,
                                float* __restrict__ vout) {
    __shared__ float sF[CH];
    __shared__ float sQb[CH];
    __shared__ float sKb[CH];
    const int row = blockIdx.x;          // b*NN + n
    const int t   = threadIdx.x;         // 0..63

    sF[t] = feat[row * CH + t];
    __syncthreads();

    float accq = bq[t], acck = bk[t], accv = bv[t];
    #pragma unroll 8
    for (int c = 0; c < CH; ++c) {
        float f = sF[c];
        accq += f * Wq[t * CH + c];
        acck += f * Wk[t * CH + c];
        accv += f * Wv[t * CH + c];
    }
    vout[row * CH + t] = accv;

    sQb[t] = accq; sKb[t] = acck;
    __syncthreads();
    float s1 = 0.f, s2 = 0.f, k1 = 0.f, k2 = 0.f;
    #pragma unroll 8
    for (int c = 0; c < CH; ++c) {
        float a = sQb[c]; s1 += a; s2 += a * a;
        float b = sKb[c]; k1 += b; k2 += b * b;
    }
    const float mq = s1 * (1.f / CH), vq = s2 * (1.f / CH) - mq * mq;
    const float mk = k1 * (1.f / CH), vk = k2 * (1.f / CH) - mk * mk;
    float qv = fmaxf(0.f, (accq - mq) * rsqrtf(vq + 1e-5f) * gq[t] + betaq[t]);
    float kv = fmaxf(0.f, (acck - mk) * rsqrtf(vk + 1e-5f) * gk[t] + betak[t]);
    __syncthreads();
    sQb[t] = qv; sKb[t] = kv;
    __syncthreads();

    float aq = 0.f, ak = b1[t];
    #pragma unroll 8
    for (int h = 0; h < CH; ++h) {
        float w = W1[t * CH + h];
        aq += sQb[h] * w;
        ak += sKb[h] * w;
    }
    qW1[row * CH + t] = aq;
    kW1[row * CH + t] = ak;
}

// W2 -> bf16 repack (row-major; B-operand loads become contiguous 32B reads)
__global__ void w2_bf16_kernel(const float* __restrict__ W2,
                               unsigned short* __restrict__ W2bf) {
    int i = blockIdx.x * blockDim.x + threadIdx.x;
    if (i < CH * CH) W2bf[i] = f2bf(W2[i]);
}

// ---------------------------------------------------------------------------
// Main fused kernel: 8 waves/block, wave w handles query i = blk*8+w.
// Double-buffered j-tiles prefetched with GLOBAL_LOAD_ASYNC_TO_LDS_B128
// (s_wait_asynccnt), LayerNorm via shfl_xor(16) lane-pair reduction, native
// bf16 A-pack, 8x v_wmma_f32_16x16x32_bf16 vs register-resident W2, online
// distribution-weighted softmax + value accumulation per channel.
// ---------------------------------------------------------------------------
__global__ void __launch_bounds__(256)
vattn_main_kernel(const float* __restrict__ qW1, const float* __restrict__ kW1,
                  const float* __restrict__ vmat, const float* __restrict__ dvec,
                  const unsigned short* __restrict__ W2bf,
                  const float* __restrict__ gw, const float* __restrict__ betaw,
                  const float* __restrict__ b2, float* __restrict__ out) {
    __shared__ float sK[2][TJ * CH];
    __shared__ float sV[2][TJ * CH];
    __shared__ float sGB[2 * CH];      // interleaved {gw[c], betaw[c]}

    const int tid  = threadIdx.x;
    const int wave = tid >> 5;
    const int lane = tid & 31;
    const int half = lane >> 4;        // lane-half: channel groups / D rows
    const int lcol = lane & 15;        // A row / D column within tile

    const int b = blockIdx.x / (NN / WAVES);
    const int i = (blockIdx.x % (NN / WAVES)) * WAVES + wave;

    if (tid < CH) { sGB[2 * tid] = gw[tid]; sGB[2 * tid + 1] = betaw[tid]; }

    // --- loop-invariant per-lane constants --------------------------------
    const float* qrow = qW1 + (size_t)(b * NN + i) * CH;
    float qw[4][8];
    #pragma unroll
    for (int g = 0; g < 4; ++g) {
        const int base = (g >> 1) * 32 + (g & 1) * 8 + half * 16;
        #pragma unroll
        for (int e = 0; e < 8; ++e) qw[g][e] = qrow[base + e];
    }
    float b2v[4];
    #pragma unroll
    for (int t = 0; t < 4; ++t) b2v[t] = b2[t * 16 + lcol];
    const float dist_i = dvec[b * NN + i];

    // B-operands: B[k,c] = W2[c,k]; lane holds column c=t*16+lcol,
    // K = q2*32 + half*16 + e -> 16 contiguous bf16 = one 32B load.
    v16bf Bop[2][4];
    #pragma unroll
    for (int q2 = 0; q2 < 2; ++q2)
        #pragma unroll
        for (int t = 0; t < 4; ++t)
            Bop[q2][t] = *(const v16bf*)(W2bf + (t * 16 + lcol) * CH + q2 * 32 + half * 16);

    float mrun[4], srun[4], arun[4];
    #pragma unroll
    for (int t = 0; t < 4; ++t) { mrun[t] = -3.0e38f; srun[t] = 0.f; arun[t] = 0.f; }

    const float* kbase = kW1  + (size_t)b * NN * CH;
    const float* vbase = vmat + (size_t)b * NN * CH;
    const float* dbase = dvec + (size_t)b * NN;

#if HAVE_ASYNC_LDS
    // prefetch tile 0 into buffer 0 (one b128 async instr per wave per array)
    async_copy16(kbase + tid * 4, &sK[0][tid * 4]);
    async_copy16(vbase + tid * 4, &sV[0][tid * 4]);
#endif
    __syncthreads();                    // covers sGB init (and nothing else)

    const int NT = NN / TJ;
    for (int tile = 0; tile < NT; ++tile) {
        const int cur = tile & 1;
        const int j0  = tile * TJ;

#if HAVE_ASYNC_LDS
        if (tile + 1 < NT) {            // prefetch next tile into other buffer
            async_copy16(kbase + (j0 + TJ) * CH + tid * 4, &sK[cur ^ 1][tid * 4]);
            async_copy16(vbase + (j0 + TJ) * CH + tid * 4, &sV[cur ^ 1][tid * 4]);
            wait_asynccnt<2>();         // current tile's 2 async ops done
        } else {
            wait_asynccnt<0>();
        }
#else
        ((float4*)sK[cur])[tid] = ((const float4*)(kbase + j0 * CH))[tid];
        ((float4*)sV[cur])[tid] = ((const float4*)(vbase + j0 * CH))[tid];
#endif
        __syncthreads();                // all waves' current-tile data landed

        const float* sKc = sK[cur];
        const float* sVc = sV[cur];

        // pre = kW1[j] - qW1[i] at this lane's 32 channels of row r=lcol
        float pre[4][8];
        float s1 = 0.f, s2 = 0.f;
        #pragma unroll
        for (int g = 0; g < 4; ++g) {
            const int base = (g >> 1) * 32 + (g & 1) * 8 + half * 16;
            const float4* p = (const float4*)(&sKc[lcol * CH + base]);
            float4 x0 = p[0], x1 = p[1];
            float vals[8] = {x0.x, x0.y, x0.z, x0.w, x1.x, x1.y, x1.z, x1.w};
            #pragma unroll
            for (int e = 0; e < 8; ++e) {
                float pv = vals[e] - qw[g][e];
                pre[g][e] = pv; s1 += pv; s2 += pv * pv;
            }
        }
        s1 += __shfl_xor(s1, 16, 32);   // lane pair holds complementary halves
        s2 += __shfl_xor(s2, 16, 32);
        const float mean = s1 * (1.f / CH);
        const float var  = s2 * (1.f / CH) - mean * mean;
        const float rinv = rsqrtf(var + 1e-5f);

        // LN -> relu -> native bf16 cast, packed in 16-bit A-matrix layout
        union { v16bf v; __bf16 h[16]; } A[2];
        #pragma unroll
        for (int g = 0; g < 4; ++g) {
            const int base = (g >> 1) * 32 + (g & 1) * 8 + half * 16;
            const int q2 = g >> 1, gb = g & 1;
            const float4* gbp = (const float4*)(&sGB[2 * base]);
            float4 c0 = gbp[0], c1 = gbp[1], c2 = gbp[2], c3 = gbp[3];
            float gwa[8] = {c0.x, c0.z, c1.x, c1.z, c2.x, c2.z, c3.x, c3.z};
            float bea[8] = {c0.y, c0.w, c1.y, c1.w, c2.y, c2.w, c3.y, c3.w};
            #pragma unroll
            for (int e = 0; e < 8; ++e) {
                float hv = fmaxf(0.f, (pre[g][e] - mean) * rinv * gwa[e] + bea[e]);
                A[q2].h[gb * 8 + e] = (__bf16)hv;
            }
        }

        // distribution weights for this lane's 8 rows (L0-resident)
        const float4* dp = (const float4*)(&dbase[j0 + half * 8]);
        float4 d0 = dp[0], d1 = dp[1];
        float dj[8] = {d0.x, d0.y, d0.z, d0.w, d1.x, d1.y, d1.z, d1.w};

        // hidden[16x64] @ W2^T[64x64] -> 4 column tiles, K=64 in 2 WMMAs each
        #pragma unroll
        for (int t = 0; t < 4; ++t) {
            v8f acc = {};
            acc = __builtin_amdgcn_wmma_f32_16x16x32_bf16(false, A[0].v, false, Bop[0][t],
                                                          (short)0, acc, false, false);
            acc = __builtin_amdgcn_wmma_f32_16x16x32_bf16(false, A[1].v, false, Bop[1][t],
                                                          (short)0, acc, false, false);
            const int ccol = t * 16 + lcol;
            const float bb = b2v[t];
            float tmax = acc[0];
            #pragma unroll
            for (int r = 1; r < 8; ++r) tmax = fmaxf(tmax, acc[r]);
            tmax += bb;
            const float mnew  = fmaxf(mrun[t], tmax);
            const float scale = __expf(mrun[t] - mnew);
            float sacc = srun[t] * scale;
            float aacc = arun[t] * scale;
            #pragma unroll
            for (int r = 0; r < 8; ++r) {
                float w = acc[r] + bb;
                float e = __expf(w - mnew) * dj[r];
                sacc += e;
                aacc += e * sVc[(r + half * 8) * CH + ccol];
            }
            mrun[t] = mnew; srun[t] = sacc; arun[t] = aacc;
        }

        __syncthreads();                // buffer reused for prefetch next iter
    }

    // merge lane-pair partials (rows 0-7 vs 8-15 of every tile) and emit
    const float outmask = dist_i > 0.f ? 1.f : 0.f;
    #pragma unroll
    for (int t = 0; t < 4; ++t) {
        float m2 = __shfl_xor(mrun[t], 16, 32);
        float s2 = __shfl_xor(srun[t], 16, 32);
        float a2 = __shfl_xor(arun[t], 16, 32);
        float M  = fmaxf(mrun[t], m2);
        float e1 = __expf(mrun[t] - M), e2 = __expf(m2 - M);
        float stot = srun[t] * e1 + s2 * e2;
        float atot = arun[t] * e1 + a2 * e2;
        if (half == 0)
            out[(size_t)(b * NN + i) * CH + t * 16 + lcol] = outmask * atot / stot;
    }
}

extern "C" void kernel_launch(void* const* d_in, const int* in_sizes, int n_in,
                              void* d_out, int out_size, void* d_ws, size_t ws_size,
                              hipStream_t stream) {
    const float* feat  = (const float*)d_in[0];
    const float* distv = (const float*)d_in[1];
    const float* Wq    = (const float*)d_in[2];
    const float* bq    = (const float*)d_in[3];
    const float* gq    = (const float*)d_in[4];
    const float* betaq = (const float*)d_in[5];
    const float* Wk    = (const float*)d_in[6];
    const float* bk    = (const float*)d_in[7];
    const float* gk    = (const float*)d_in[8];
    const float* betak = (const float*)d_in[9];
    const float* Wv    = (const float*)d_in[10];
    const float* bv    = (const float*)d_in[11];
    const float* W1    = (const float*)d_in[12];
    const float* b1    = (const float*)d_in[13];
    const float* gw    = (const float*)d_in[14];
    const float* betaw = (const float*)d_in[15];
    const float* W2    = (const float*)d_in[16];
    const float* b2    = (const float*)d_in[17];

    char* ws = (char*)d_ws;
    const size_t rowsz = (size_t)BB * NN * CH * sizeof(float);   // 512 KB
    float* qW1  = (float*)(ws);
    float* kW1  = (float*)(ws + rowsz);
    float* vmat = (float*)(ws + 2 * rowsz);
    unsigned short* W2bf = (unsigned short*)(ws + 3 * rowsz);    // 32B-aligned

    hipLaunchKernelGGL(qkv_prep_kernel, dim3(BB * NN), dim3(CH), 0, stream,
                       feat, Wq, bq, gq, betaq, Wk, bk, gk, betak, Wv, bv, W1, b1,
                       qW1, kW1, vmat);
    hipLaunchKernelGGL(w2_bf16_kernel, dim3((CH * CH + 255) / 256), dim3(256), 0, stream,
                       W2, W2bf);
    hipLaunchKernelGGL(vattn_main_kernel, dim3(BB * NN / WAVES), dim3(256), 0, stream,
                       qW1, kW1, vmat, distv, W2bf, gw, betaw, b2, (float*)d_out);
}
